// GatedGNN_12987981103195
// MI455X (gfx1250) — matrix operand order repaired and checked by reading, not compiled, not run on hardware
//
#include <hip/hip_runtime.h>
#include <hip/hip_bf16.h>
#include <math.h>

typedef __bf16 bf16;
typedef __attribute__((ext_vector_type(8)))  __bf16 bf16x8;
typedef __attribute__((ext_vector_type(16))) __bf16 bf16x16;
typedef __attribute__((ext_vector_type(8)))  float  f32x8;
typedef __attribute__((ext_vector_type(8)))  int    i32x8;

#define WMMA_BF16(A,B,C) __builtin_amdgcn_wmma_f32_16x16x32_bf16( \
    false,(A),false,(B),(short)0,(C),false,false)

constexpr int Bsz = 8, Nn = 2048, Dd = 256, Hh = 256;
constexpr int BN  = Bsz * Nn;                // 16384 flattened rows

// ---------------------------------------------------------------------------
// Elementwise f32 -> bf16 convert (x -> x_bf16, row major)
// ---------------------------------------------------------------------------
__global__ __launch_bounds__(256) void k_cvt(const float* __restrict__ src,
                                             bf16* __restrict__ dst, int n) {
    int i = blockIdx.x * 256 + threadIdx.x;
    if (i < n) dst[i] = (bf16)src[i];
}

// ---------------------------------------------------------------------------
// Pack a (K x Ncols) f32 matrix into the WMMA bf16 B-operand lane layout.
// Packed tile (kt, nt): 32 lanes x 32 bytes contiguous. Lane l owns column
// nt*16 + (l&15), K-rows kt*32 + (l>>4)*8 + {0..7, 16..23} packed 2-per-dword.
// trans==1 means logical B[k,col] = src[col*K + k] (i.e. B = src^T).
// ---------------------------------------------------------------------------
__global__ __launch_bounds__(256) void k_pack(const float* __restrict__ src,
                                              bf16* __restrict__ dst,
                                              int K, int Ncols, int trans, int total) {
    int idx = blockIdx.x * 256 + threadIdx.x;
    if (idx >= total) return;
    int t    = idx >> 9;            // 512 bf16 elements per 32x16 tile
    int r    = idx & 511;
    int lane = r >> 4;
    int d    = (r >> 1) & 7;
    int p    = r & 1;
    int ntn  = Ncols >> 4;
    int kt   = t / ntn, nt = t % ntn;
    int k    = kt * 32 + (lane >> 4) * 8 + ((d & 4) << 2) + (d & 3) * 2 + p;
    int col  = nt * 16 + (lane & 15);
    float v  = trans ? src[(size_t)col * K + k] : src[(size_t)k * Ncols + col];
    dst[idx] = (bf16)v;
}

// ---------------------------------------------------------------------------
// K1: Wh = x @ W  (bf16 WMMA, f32 accum), stored transposed: WhT[b][h][i]
// grid = BN/16 blocks, 128 threads (4 waves). Wave w covers h in [64w,64w+64).
// ---------------------------------------------------------------------------
__global__ __launch_bounds__(128) void k_wh(const bf16* __restrict__ xb,
                                            const bf16* __restrict__ Wp,
                                            bf16* __restrict__ WhT) {
    int blk  = blockIdx.x;
    int b    = blk / (Nn / 16);
    int i0   = (blk % (Nn / 16)) * 16;
    int lane = threadIdx.x & 31;
    int w    = threadIdx.x >> 5;
    int row  = lane & 15;
    int hi   = lane >> 4;

    f32x8 acc[4] = {};
    const bf16* arow = xb + (size_t)(b * Nn + i0 + row) * Dd + hi * 8;

    for (int kt = 0; kt < Dd / 32; ++kt) {
        bf16x8 lo  = *(const bf16x8*)(arow + kt * 32);
        bf16x8 hi8 = *(const bf16x8*)(arow + kt * 32 + 16);
        bf16x16 A;
#pragma unroll
        for (int q = 0; q < 8; ++q) { A[q] = lo[q]; A[8 + q] = hi8[q]; }
#pragma unroll
        for (int ht = 0; ht < 4; ++ht) {
            int nt = w * 4 + ht;
            bf16x16 Bv = *(const bf16x16*)(Wp + ((size_t)(kt * 16 + nt) * 32 + lane) * 16);
            acc[ht] = WMMA_BF16(A, Bv, acc[ht]);
        }
    }
    // D tile: lane holds col = l&15, rows i0 + (l>>4)*8 + v  -> contiguous i
    // in WhT, so each lane emits one 16B vector store per tile.
#pragma unroll
    for (int ht = 0; ht < 4; ++ht) {
        int hcol = w * 64 + ht * 16 + row;
        bf16x8 o;
#pragma unroll
        for (int v = 0; v < 8; ++v) o[v] = (bf16)acc[ht][v];
        *(bf16x8*)(WhT + (size_t)(b * Hh + hcol) * Nn + i0 + hi * 8) = o;
    }
}

// ---------------------------------------------------------------------------
// K1b: Wh1[i] = sum_h Wh[i,h]*a[h], Wh2[i] = sum_h Wh[i,h]*a[H+h]
// Reads WhT: for fixed h, consecutive i are contiguous -> coalesced.
// ---------------------------------------------------------------------------
__global__ __launch_bounds__(256) void k_wh12(const bf16* __restrict__ WhT,
                                              const float* __restrict__ a,
                                              float* __restrict__ Wh1,
                                              float* __restrict__ Wh2) {
    int i  = blockIdx.x * 256 + threadIdx.x;   // 0..BN-1
    int b  = i / Nn, ii = i % Nn;
    const bf16* base = WhT + (size_t)b * Hh * Nn + ii;
    float s1 = 0.f, s2 = 0.f;
    for (int h = 0; h < Hh; ++h) {
        float v = (float)base[(size_t)h * Nn];
        s1 += v * a[h];
        s2 += v * a[Hh + h];
    }
    Wh1[i] = s1;
    Wh2[i] = s2;
}

// ---------------------------------------------------------------------------
// K2: fused masked-softmax attention (flash style).
// graph_out[b,i,:] = softmax_m(mask(leaky(Wh1_i+Wh2_m))) @ Wh[b,:,:]
// grid = B*(N/16), 128 threads; each wave independently owns 64 h-columns
// and redundantly computes the shared 16x32 P tile (A operand) per m-block.
// ---------------------------------------------------------------------------
__global__ __launch_bounds__(128) void k_attn(const int*  __restrict__ adj,
                                              const float* __restrict__ Wh1,
                                              const float* __restrict__ Wh2,
                                              const bf16* __restrict__ WhT,
                                              bf16* __restrict__ go) {
    int blk  = blockIdx.x;
    int b    = blk / (Nn / 16);
    int i0   = (blk % (Nn / 16)) * 16;
    int lane = threadIdx.x & 31;
    int w    = threadIdx.x >> 5;
    int row  = lane & 15;
    int hi   = lane >> 4;
    const float NEG = -1.0e30f;

    float wh1r = Wh1[b * Nn + i0 + row];
    const int*   adjrow = adj + (size_t)(b * Nn + i0 + row) * Nn;
    const float* wh2b   = Wh2 + b * Nn;

    f32x8 acc[4] = {};
    float m_run = -INFINITY, s_run = 0.f;

    for (int mb = 0; mb < Nn / 32; ++mb) {
        int mbase = mb * 32 + hi * 16;
        // scores, in the contiguous k order shared with V: e[0..15] = K
        // (l>>4)*16+{0..15}.
        float e[16];
#pragma unroll
        for (int c = 0; c < 2; ++c) {
            i32x8 msk = *(const i32x8*)(adjrow + mbase + c * 8);
            f32x8 w2  = *(const f32x8*)(wh2b + mbase + c * 8);
#pragma unroll
            for (int j = 0; j < 8; ++j) {
                float ev = wh1r + w2[j];
                ev = ev > 0.f ? ev : 0.2f * ev;           // leaky_relu(0.2)
                e[c * 8 + j] = (msk[j] > 0) ? ev : NEG;
            }
        }
        // online softmax: lanes l and l+16 share row l&15
        float emax = e[0];
#pragma unroll
        for (int j = 1; j < 16; ++j) emax = fmaxf(emax, e[j]);
        emax = fmaxf(emax, __shfl_xor(emax, 16, 32));
        float m_new = fmaxf(m_run, emax);

        bf16x16 P;
        float psum = 0.f;
#pragma unroll
        for (int j = 0; j < 16; ++j) {
            float p = __expf(e[j] - m_new);
            psum += p;
            P[j] = (bf16)p;
        }
        psum += __shfl_xor(psum, 16, 32);
        float corr = __expf(m_run - m_new);
        s_run = s_run * corr + psum;
        m_run = m_new;

        // rescale accumulators; per-row factor broadcast from lanes 0..15
#pragma unroll
        for (int v = 0; v < 8; ++v) {
            float fac = __shfl(corr, hi * 8 + v, 32);
#pragma unroll
            for (int ht = 0; ht < 4; ++ht) acc[ht][v] *= fac;
        }
        // V tiles: contiguous 32B per lane from transposed Wh
        const bf16* vbase = WhT + (size_t)(b * Hh + w * 64 + row) * Nn + mb * 32 + hi * 16;
        if (mb + 1 < Nn / 32) __builtin_prefetch(vbase + 32, 0, 1);
#pragma unroll
        for (int ht = 0; ht < 4; ++ht) {
            bf16x16 V = *(const bf16x16*)(vbase + (size_t)(ht * 16) * Nn);
            acc[ht] = WMMA_BF16(P, V, acc[ht]);
        }
    }
    // normalize and store graph_out (bf16, row major)
#pragma unroll
    for (int v = 0; v < 8; ++v) {
        float inv = 1.0f / __shfl(s_run, hi * 8 + v, 32);
        int m = i0 + hi * 8 + v;
#pragma unroll
        for (int ht = 0; ht < 4; ++ht) {
            int hcol = w * 64 + ht * 16 + row;
            go[(size_t)(b * Nn + m) * Hh + hcol] = (bf16)(acc[ht][v] * inv);
        }
    }
}

// ---------------------------------------------------------------------------
// K3: fused GRU cell. gi = go @ w_ih^T, gh = x @ w_hh^T (WMMA), then gates.
// grid = (BN/16) * (H/64), 128 threads; wave w owns 16 output columns.
// ---------------------------------------------------------------------------
__global__ __launch_bounds__(128) void k_gru(const bf16* __restrict__ gob,
                                             const bf16* __restrict__ xb,
                                             const float* __restrict__ xf,
                                             const bf16* __restrict__ Wihp,
                                             const bf16* __restrict__ Whhp,
                                             const float* __restrict__ bih,
                                             const float* __restrict__ bhh,
                                             float* __restrict__ out) {
    int blk  = blockIdx.x;
    int i0   = (blk / (Hh / 64)) * 16;
    int hc   = blk % (Hh / 64);
    int lane = threadIdx.x & 31;
    int w    = threadIdx.x >> 5;
    int row  = lane & 15;
    int hi   = lane >> 4;

    f32x8 gi[3] = {}, gh[3] = {};
    const bf16* a1 = gob + (size_t)(i0 + row) * Hh + hi * 8;
    const bf16* a2 = xb  + (size_t)(i0 + row) * Dd + hi * 8;

    for (int kt = 0; kt < 8; ++kt) {
        bf16x16 A1, A2;
        {
            bf16x8 lo = *(const bf16x8*)(a1 + kt * 32);
            bf16x8 h8 = *(const bf16x8*)(a1 + kt * 32 + 16);
#pragma unroll
            for (int q = 0; q < 8; ++q) { A1[q] = lo[q]; A1[8 + q] = h8[q]; }
        }
        {
            bf16x8 lo = *(const bf16x8*)(a2 + kt * 32);
            bf16x8 h8 = *(const bf16x8*)(a2 + kt * 32 + 16);
#pragma unroll
            for (int q = 0; q < 8; ++q) { A2[q] = lo[q]; A2[8 + q] = h8[q]; }
        }
#pragma unroll
        for (int g = 0; g < 3; ++g) {
            int nt = g * 16 + hc * 4 + w;   // column tile g*H + hc*64 + w*16
            bf16x16 B1 = *(const bf16x16*)(Wihp + ((size_t)(kt * 48 + nt) * 32 + lane) * 16);
            bf16x16 B2 = *(const bf16x16*)(Whhp + ((size_t)(kt * 48 + nt) * 32 + lane) * 16);
            gi[g] = WMMA_BF16(A1, B1, gi[g]);
            gh[g] = WMMA_BF16(A2, B2, gh[g]);
        }
    }
    int hcol = hc * 64 + w * 16 + row;
    float bir = bih[hcol], biz = bih[Hh + hcol], bin = bih[2 * Hh + hcol];
    float bhr = bhh[hcol], bhz = bhh[Hh + hcol], bhn = bhh[2 * Hh + hcol];
#pragma unroll
    for (int v = 0; v < 8; ++v) {
        int m   = i0 + hi * 8 + v;
        float r = 1.f / (1.f + __expf(-(gi[0][v] + bir + gh[0][v] + bhr)));
        float z = 1.f / (1.f + __expf(-(gi[1][v] + biz + gh[1][v] + bhz)));
        float n = tanhf(gi[2][v] + bin + r * (gh[2][v] + bhn));
        float hp = xf[(size_t)m * Dd + hcol];
        out[(size_t)m * Hh + hcol] = (1.f - z) * n + z * hp;
    }
}

// ---------------------------------------------------------------------------
extern "C" void kernel_launch(void* const* d_in, const int* in_sizes, int n_in,
                              void* d_out, int out_size, void* d_ws, size_t ws_size,
                              hipStream_t stream) {
    const int*   adj  = (const int*)  d_in[0];
    const float* x    = (const float*)d_in[1];
    const float* W    = (const float*)d_in[2];
    const float* a    = (const float*)d_in[3];
    const float* wih  = (const float*)d_in[4];
    const float* whh  = (const float*)d_in[5];
    const float* bih  = (const float*)d_in[6];
    const float* bhh  = (const float*)d_in[7];
    float* out = (float*)d_out;

    char* ws = (char*)d_ws;
    bf16*  xb   = (bf16*)ws;  ws += (size_t)BN * Dd * 2;       // 8 MB
    bf16*  WhT  = (bf16*)ws;  ws += (size_t)Bsz * Hh * Nn * 2; // 8 MB
    bf16*  gob  = (bf16*)ws;  ws += (size_t)BN * Hh * 2;       // 8 MB
    float* Wh1  = (float*)ws; ws += (size_t)BN * 4;
    float* Wh2  = (float*)ws; ws += (size_t)BN * 4;
    bf16*  Wp   = (bf16*)ws;  ws += (size_t)Dd * Hh * 2;
    bf16*  Wihp = (bf16*)ws;  ws += (size_t)Dd * 3 * Hh * 2;
    bf16*  Whhp = (bf16*)ws;  ws += (size_t)Hh * 3 * Hh * 2;

    // stage 0: converts + weight packing
    k_cvt<<<(BN * Dd + 255) / 256, 256, 0, stream>>>(x, xb, BN * Dd);
    k_pack<<<(Dd * Hh + 255) / 256, 256, 0, stream>>>(W,   Wp,   Dd, Hh,     0, Dd * Hh);
    k_pack<<<(Dd * 3 * Hh + 255) / 256, 256, 0, stream>>>(wih, Wihp, Dd, 3 * Hh, 1, Dd * 3 * Hh);
    k_pack<<<(Hh * 3 * Hh + 255) / 256, 256, 0, stream>>>(whh, Whhp, Hh, 3 * Hh, 1, Hh * 3 * Hh);
    // stage 1: Wh = x @ W (transposed store) and attention logit vectors
    k_wh  <<<BN / 16, 128, 0, stream>>>(xb, Wp, WhT);
    k_wh12<<<BN / 256, 256, 0, stream>>>(WhT, a, Wh1, Wh2);
    // stage 2: fused masked softmax attention (flash style)
    k_attn<<<BN / 16, 128, 0, stream>>>(adj, Wh1, Wh2, WhT, gob);
    // stage 3: fused GRU cell -> final output
    k_gru <<<(BN / 16) * (Hh / 64), 128, 0, stream>>>(gob, xb, x, Wihp, Whhp, bih, bhh, out);
}
